// AudioSNN_20976620273875
// MI455X (gfx1250) — compile-verified, matrix operand order checked by reading, not run on hardware
//
#include <hip/hip_runtime.h>

// ---- problem constants (from reference) ----
#define B_TOT   4096
#define T_STEPS 500
#define F_IN    40
#define H_HID   128
#define O_OUT   5
#define BETA    0.9f
#define THR     1.0f

typedef __attribute__((ext_vector_type(16))) _Float16     v16h;
typedef __attribute__((ext_vector_type(8)))  _Float16     v8h;
typedef __attribute__((ext_vector_type(8)))  float        v8f;
typedef __attribute__((ext_vector_type(4)))  float        v4f;
typedef __attribute__((ext_vector_type(4)))  unsigned int v4u;
typedef __attribute__((ext_vector_type(8)))  int          v8i;
typedef __attribute__((ext_vector_type(4)))  int          v4i;

// Convert 8 f32 (two b128 LDS loads) to 8 packed f16 halves.
__device__ __forceinline__ v8h cvt8(v4f a, v4f b) {
    v8h r;
    r[0] = (_Float16)a[0]; r[1] = (_Float16)a[1];
    r[2] = (_Float16)a[2]; r[3] = (_Float16)a[3];
    r[4] = (_Float16)b[0]; r[5] = (_Float16)b[1];
    r[6] = (_Float16)b[2]; r[7] = (_Float16)b[3];
    return r;
}

// A-matrix (16x32 f16) fragment gather from row-major f16 [m][k] LDS tile (contiguous K).
__device__ __forceinline__ v16h load_frag_A16(const _Float16* row, int kbase, int lh) {
    v8h lo = *(const v8h*)(row + kbase + lh * 8);
    v8h hi = *(const v8h*)(row + kbase + 16 + lh * 8);
    return __builtin_shufflevector(lo, hi, 0,1,2,3,4,5,6,7,8,9,10,11,12,13,14,15);
}

// B-matrix (32x16 f16) fragment gather from [k][n] LDS layout; weights only (pre-loop).
__device__ __forceinline__ v16h load_frag_B(const _Float16* base, int kbase, int stride,
                                            int lh, int n) {
    v16h r;
#pragma unroll
    for (int h = 0; h < 8; ++h)
        r[h] = base[(kbase + lh * 8 + h) * stride + n];
#pragma unroll
    for (int h = 0; h < 8; ++h)
        r[8 + h] = base[(kbase + 16 + lh * 8 + h) * stride + n];
    return r;
}

// 128 threads = 4 wave32. Workgroup owns 16 batch rows for all 500 timesteps.
// wave w computes hidden tiles {2w,2w+1} of layer 1 (m1 in WMMA accumulators);
// wave 0 owns the TDM pipeline and layer 2 (m2), and writes spk2.
__global__ __launch_bounds__(128) void snn_scan_kernel(
    const float* __restrict__ x,  const float* __restrict__ W1,
    const float* __restrict__ b1, const float* __restrict__ W2,
    const float* __restrict__ b2, float* __restrict__ out)
{
    __shared__ _Float16 w1lds[64 * H_HID];                  // [k][n]=W1^T, K pad 40->64 (16KB)
    __shared__ _Float16 w2lds[H_HID * 16];                  // [k][n]=W2^T, N pad 5->16   (4KB)
    __shared__ _Float16 spk1s[16 * H_HID];                  // [m][n] layer-1 spikes      (4KB)
    __shared__ float __attribute__((aligned(16)))
        xtile[2][16 * F_IN];                                // raw f32 TDM tiles          (5KB)

    const int tid  = threadIdx.x;
    const int wave = tid >> 5;
    const int lane = tid & 31;
    const int ln   = lane & 15;   // N (or A-row M) index within half-wave
    const int lh   = lane >> 4;   // half select
    const int b0   = blockIdx.x * 16;
    const size_t strideB = (size_t)T_STEPS * F_IN;          // x batch-row stride (elems)

    // ---- stage time-invariant weights into LDS (transposed to [k][n], padded) ----
    for (int i = tid; i < 64 * H_HID; i += 128) {
        int k = i / H_HID, n = i % H_HID;
        w1lds[i] = (k < F_IN) ? (_Float16)W1[n * F_IN + k] : (_Float16)0.0f;
    }
    for (int i = tid; i < H_HID * 16; i += 128) {
        int k = i / 16, n = i % 16;
        w2lds[i] = (n < O_OUT) ? (_Float16)W2[n * H_HID + k] : (_Float16)0.0f;
    }

    // ---- TDM descriptor: 2D tile 16 rows x 40 f32, row stride T*F elems ----
    // Group1 is time-invariant; group0 carries lds/global addresses per issue.
    v8i g1;
    g1[0] = 0x00020000;          // workgroup_mask=0, data_size=2 (4 bytes)
    g1[1] = (int)(40u << 16);    // tensor_dim0[15:0]=40
    g1[2] = (int)(16u << 16);    // tensor_dim0 hi=0, tensor_dim1[15:0]=16
    g1[3] = (int)(40u << 16);    // tensor_dim1 hi=0, tile_dim0=40
    g1[4] = 16;                  // tile_dim1=16, tile_dim2=0
    g1[5] = (int)strideB;        // tensor_dim0_stride lo32 = 20000
    g1[6] = 0;                   // stride hi, tensor_dim1_stride lo
    g1[7] = 0;
    const v4i gz4 = {0, 0, 0, 0};
    const v8i gz8 = {0, 0, 0, 0, 0, 0, 0, 0};
    const unsigned ldsA[2] = { (unsigned)(uintptr_t)&xtile[0][0],
                               (unsigned)(uintptr_t)&xtile[1][0] };

    auto issue_tdm = [&](int tt) {
        unsigned long long ga =
            (unsigned long long)(uintptr_t)(x + (size_t)b0 * strideB + (size_t)tt * F_IN);
        v4u g0;
        g0[0] = 1u;                                          // count=1 (valid descriptor)
        g0[1] = ldsA[tt & 1];                                // lds_addr
        g0[2] = (unsigned)ga;                                // global_addr[31:0]
        g0[3] = (unsigned)((ga >> 32) & 0x01FFFFFFu) | (2u << 30);  // ga[56:32], type=2
        // 6-arg clang-23 form: (g0, g1, g2, g3, g4, cpol); 2D tensor -> extra groups zero
        __builtin_amdgcn_tensor_load_to_lds(g0, g1, gz4, gz4, gz8, 0);
    };

    // ---- per-lane biases and invariant B fragments (resident for the whole scan) ----
    const int j0 = 2 * wave;
    float bias1a = b1[(j0 + 0) * 16 + ln];
    float bias1b = b1[(j0 + 1) * 16 + ln];
    float bias2  = (ln < O_OUT) ? b2[ln] : 0.0f;

    __syncthreads();             // weights staged
    v16h B1[2][2];
#pragma unroll
    for (int jj = 0; jj < 2; ++jj)
#pragma unroll
        for (int kt = 0; kt < 2; ++kt)
            B1[jj][kt] = load_frag_B(w1lds, kt * 32, H_HID, lh, (j0 + jj) * 16 + ln);
    v16h B2[4];
#pragma unroll
    for (int kt = 0; kt < 4; ++kt)
        B2[kt] = load_frag_B(w2lds, kt * 32, 16, lh, ln);

    v8f m1[2] = {{0.f,0.f,0.f,0.f,0.f,0.f,0.f,0.f}, {0.f,0.f,0.f,0.f,0.f,0.f,0.f,0.f}};
    v8f m2    =  {0.f,0.f,0.f,0.f,0.f,0.f,0.f,0.f};

    // ---- prologue: start the DMA for t=0 ----
    if (wave == 0) issue_tdm(0);

    for (int t = 0; t < T_STEPS; ++t) {
        // pipeline: kick DMA for t+1, retire DMA for t (TDM ops are in-order per wave)
        if (wave == 0) {
            if (t + 1 < T_STEPS) {
                issue_tdm(t + 1);
                __builtin_amdgcn_s_wait_tensorcnt(1);
            } else {
                __builtin_amdgcn_s_wait_tensorcnt(0);
            }
        }
        if (t + 2 < T_STEPS && tid < 16)   // keep HBM stream ahead of the TDM
            __builtin_prefetch(&x[(size_t)(b0 + tid) * strideB + (size_t)(t + 2) * F_IN], 0, 3);
        __syncthreads();                    // tile t visible to all waves

        // ---- layer 1: cur1 = xt @ W1^T ; m1 = beta*m1 + cur1 + b1 - H(m1-thr) ----
        const float* row = &xtile[t & 1][0] + ln * F_IN;   // this lane's M row (40 f32)
        // fragment kt=0: K=0..31 (all valid)
        v4f p0 = *(const v4f*)(row + lh * 8);
        v4f p1 = *(const v4f*)(row + lh * 8 + 4);
        v4f p2 = *(const v4f*)(row + 16 + lh * 8);
        v4f p3 = *(const v4f*)(row + 16 + lh * 8 + 4);
        v16h A0 = __builtin_shufflevector(cvt8(p0, p1), cvt8(p2, p3),
                                          0,1,2,3,4,5,6,7,8,9,10,11,12,13,14,15);
        // fragment kt=1: only K=32..39 valid (lh==0 lanes); rest is zero padding
        v16h A1;
#pragma unroll
        for (int h = 0; h < 16; ++h) A1[h] = (_Float16)0.0f;
        if (lh == 0) {
            v4f q0 = *(const v4f*)(row + 32);
            v4f q1 = *(const v4f*)(row + 36);
            v8h lo = cvt8(q0, q1);
#pragma unroll
            for (int h = 0; h < 8; ++h) A1[h] = lo[h];
        }

#pragma unroll
        for (int jj = 0; jj < 2; ++jj) {
            float bias = jj ? bias1b : bias1a;
            v8f c;
#pragma unroll
            for (int e = 0; e < 8; ++e) {
                float mp = m1[jj][e];
                c[e] = BETA * mp + bias - ((mp > THR) ? THR : 0.0f);
            }
            c = __builtin_amdgcn_wmma_f32_16x16x32_f16(false, A0, false, B1[jj][0],
                                                       (short)0, c, false, false);
            c = __builtin_amdgcn_wmma_f32_16x16x32_f16(false, A1, false, B1[jj][1],
                                                       (short)0, c, false, false);
            m1[jj] = c;
            // spk1 -> LDS: C layout element e is (M = e + 8*lh, N = ln) of this tile
#pragma unroll
            for (int e = 0; e < 8; ++e)
                spk1s[(e + 8 * lh) * H_HID + (j0 + jj) * 16 + ln] =
                    (c[e] > THR) ? (_Float16)1.0f : (_Float16)0.0f;
        }
        __syncthreads();                    // spk1 complete

        // ---- layer 2 (wave 0 only; EXEC full within the wave) ----
        if (wave == 0) {
            v8f c;
#pragma unroll
            for (int e = 0; e < 8; ++e) {
                float mp = m2[e];
                c[e] = BETA * mp + bias2 - ((mp > THR) ? THR : 0.0f);
            }
            const _Float16* srow = spk1s + ln * H_HID;
#pragma unroll
            for (int kt = 0; kt < 4; ++kt) {
                v16h As = load_frag_A16(srow, kt * 32, lh);
                c = __builtin_amdgcn_wmma_f32_16x16x32_f16(false, As, false, B2[kt],
                                                           (short)0, c, false, false);
            }
            m2 = c;
            if (ln < O_OUT) {
#pragma unroll
                for (int e = 0; e < 8; ++e) {
                    int m = e + 8 * lh;
                    out[(size_t)t * B_TOT * O_OUT + (size_t)(b0 + m) * O_OUT + ln] =
                        (c[e] > THR) ? 1.0f : 0.0f;
                }
            }
        }
        __syncthreads();                    // protect spk1s / tile buffers for next step
    }
}

extern "C" void kernel_launch(void* const* d_in, const int* in_sizes, int n_in,
                              void* d_out, int out_size, void* d_ws, size_t ws_size,
                              hipStream_t stream) {
    (void)in_sizes; (void)n_in; (void)d_ws; (void)ws_size; (void)out_size;
    const float* x  = (const float*)d_in[0];
    const float* W1 = (const float*)d_in[1];
    const float* b1 = (const float*)d_in[2];
    const float* W2 = (const float*)d_in[3];
    const float* b2 = (const float*)d_in[4];
    float* out = (float*)d_out;

    dim3 grid(B_TOT / 16);   // 256 workgroups x 4 waves = 1024 resident waves
    dim3 block(128);
    snn_scan_kernel<<<grid, block, 0, stream>>>(x, W1, b1, W2, b2, out);
}
